// SPD_DCNv2Down_22058952032385
// MI455X (gfx1250) — matrix operand order repaired and checked by reading, not compiled, not run on hardware
//
#include <hip/hip_runtime.h>
#include <hip/hip_bf16.h>
#include <math.h>

typedef _Float16 v16h __attribute__((ext_vector_type(16)));
typedef float    v8f  __attribute__((ext_vector_type(8)));
typedef unsigned int u32x4 __attribute__((ext_vector_type(4)));
typedef int          i32x4 __attribute__((ext_vector_type(4)));
typedef int          i32x8 __attribute__((ext_vector_type(8)));

#define EPS_ 1e-5f

// Problem dimensions (fixed by the reference).
constexpr int B_   = 8;
constexpr int C1   = 128;
constexpr int H0   = 160, W0 = 160;
constexpr int CSPD = 512;           // C1*4 after space-to-depth
constexpr int CMID = 128;
constexpr int C2   = 256;
constexpr int H    = 80, W = 80, HW = 80 * 80;   // 6400
constexpr int NPIX = B_ * HW;                     // 51200
constexpr int KDCN = 9 * CMID;                    // 1152
constexpr int KSTEPS = KDCN / 32;                 // 36

__device__ __forceinline__ float dsigmoid(float y) { return 1.0f / (1.0f + __expf(-y)); }
__device__ __forceinline__ float bnsilu(float x, float g, float be, float mu, float va) {
    float s = g * rsqrtf(va + EPS_);
    float y = (x - mu) * s + be;
    return y * dsigmoid(y);
}

// 32-byte fragment container for one wave32 WMMA 16-bit operand (8 VGPRs).
union Frag16 { uint4 u[2]; v16h h; };

// A-matrix 16x32 f16 fragment per CDNA5 ISA layout (A stored row-major,
// 32 K-halfs per row, rows 64B):
//   lane L: m = L&15, hf = L>>4
//   v16h elems 0..7  -> K = 8*hf + 0..7      (byte off 16*hf)
//   v16h elems 8..15 -> K = 16 + 8*hf + 0..7 (byte off 32 + 16*hf)
__device__ __forceinline__ v16h load_a_gl(const _Float16* __restrict__ A, int ldk,
                                          int m0, int k0, int lane) {
    int m = lane & 15, hf = lane >> 4;
    const _Float16* row = A + (size_t)(m0 + m) * ldk + k0 + 8 * hf;
    Frag16 f;
    f.u[0] = *(const uint4*)(row);
    f.u[1] = *(const uint4*)(row + 16);
    return f.h;
}
__device__ __forceinline__ v16h load_a_lds(const _Float16 (*__restrict__ Ash)[32],
                                           int m0, int lane) {
    int m = lane & 15, hf = lane >> 4;
    const _Float16* row = &Ash[m0 + m][8 * hf];
    Frag16 f;
    f.u[0] = *(const uint4*)(row);
    f.u[1] = *(const uint4*)(row + 16);
    return f.h;
}

// B-matrix 32x16 f16 fragment (same K order as the A fragment):
//   lane L: n = L&15, hf = L>>4; elems 0..7 -> K = 8*hf + 0..7, elems 8..15 -> K = 16 + 8*hf + 0..7
// Source stored pixel-major: row n holds 32 K-halfs (64B).
__device__ __forceinline__ v16h load_b_gl(const _Float16* __restrict__ X,
                                          int pbase, int k0, int lane) {
    int n = lane & 15, hf = lane >> 4;
    const _Float16* ptr = X + (size_t)(pbase + n) * CSPD + k0 + 8 * hf;
    Frag16 f;
    f.u[0] = *(const uint4*)(ptr);
    f.u[1] = *(const uint4*)(ptr + 16);
    return f.h;
}
__device__ __forceinline__ v16h load_b_lds(const _Float16 (*__restrict__ Bsh)[32],
                                           int n0, int lane) {
    int n = n0 + (lane & 15), hf = lane >> 4;
    const _Float16* ptr = &Bsh[n][8 * hf];
    Frag16 f;
    f.u[0] = *(const uint4*)(ptr);
    f.u[1] = *(const uint4*)(ptr + 16);
    return f.h;
}

// ---------------------------------------------------------------------------
// Tensor Data Mover: DMA one 2D f16 slab [rows=256][k=32] (row stride KDCN
// halfs) from global into LDS. D# built per CDNA5 ISA ch.8 bit layout.
// This toolchain exposes the 6-arg builtin:
//   (u32x4 g0, i32x8 g1, i32x4 g2, i32x4 g3, i32x8 g4, i32 cpol)
// For a 2D tile, groups 2/3 (and the trailing group) are unused -> zeros.
// ---------------------------------------------------------------------------
#if __has_builtin(__builtin_amdgcn_tensor_load_to_lds)
#define HAVE_TDM 1
__device__ __forceinline__ void tdm_load_a_slab(const _Float16* gsrc, unsigned lds_addr) {
    unsigned long long ga = (unsigned long long)(uintptr_t)gsrc;
    u32x4 g0;
    g0[0] = 1u;                                      // count=1 (valid), load, no gather
    g0[1] = lds_addr;                                // D#.lds_addr
    g0[2] = (unsigned)(ga & 0xffffffffu);            // global_addr[31:0]
    g0[3] = (unsigned)((ga >> 32) & 0x01ffffffu)     // global_addr[56:32]
          | (2u << 30);                              // type = 2 ("image")
    i32x8 g1;
    g1[0] = (int)(1u << 16);                         // data_size=1 -> 2 bytes/elem
    g1[1] = (int)((unsigned)KDCN << 16);             // tensor_dim0[15:0]
    g1[2] = (int)(256u << 16);                       // tensor_dim0 hi | tensor_dim1 lo
    g1[3] = (int)(32u << 16);                        // tensor_dim1 hi | tile_dim0=32
    g1[4] = 256;                                     // tile_dim1=256, tile_dim2=0
    g1[5] = KDCN;                                    // tensor_dim0_stride (elems)
    g1[6] = 0;
    g1[7] = 0;
    i32x4 z4 = {0, 0, 0, 0};
    i32x8 z8 = {0, 0, 0, 0, 0, 0, 0, 0};
    __builtin_amdgcn_tensor_load_to_lds(g0, g1, z4, z4, z8, 0);
}
#else
#define HAVE_TDM 0
#endif

// ---------------------------------------------------------------------------
// Kernel 0: weight precision/layout prep.
// ---------------------------------------------------------------------------
__global__ __launch_bounds__(256) void k_prep(const float* __restrict__ rw,
                                              const float* __restrict__ dw,
                                              _Float16* __restrict__ rw16,
                                              _Float16* __restrict__ dw16) {
    int i = blockIdx.x * 256 + threadIdx.x;
    if (i < CMID * CSPD) rw16[i] = (_Float16)rw[i];
    if (i < C2 * KDCN) {
        int o = i / KDCN, r = i % KDCN;
        int c = r / 9, k = r % 9;
        dw16[(size_t)o * KDCN + k * CMID + c] = (_Float16)dw[i];  // kk = tap*128+c
    }
}

// ---------------------------------------------------------------------------
// Kernel 0b: space-to-depth transpose+convert: x (f32 NCHW 160x160) ->
// Xspd16[p][cs] f16, p = pixel (b,hw), cs = SPD channel. One thread packs
// 8 consecutive cs and stores one b128. Makes GEMM1's B operand directly
// fragment-loadable from global (no LDS in the GEMM loop).
// ---------------------------------------------------------------------------
__global__ __launch_bounds__(256) void k_spd16(const float* __restrict__ x,
                                               _Float16* __restrict__ xspd) {
    int i = blockIdx.x * 256 + threadIdx.x;       // over NPIX*512/8 groups
    int p = i >> 6, g8 = i & 63;
    int cs0 = g8 * 8;
    int b = p / HW, hw = p % HW, h = hw / W, w = hw % W;
    union { uint4 u; _Float16 hv[8]; } pk;
#pragma unroll
    for (int j = 0; j < 8; ++j) {
        int cs = cs0 + j;
        int c = cs >> 2, sy = (cs >> 1) & 1, sx = cs & 1;
        size_t idx = (((size_t)b * C1 + c) * H0 + (2 * h + sy)) * W0 + (2 * w + sx);
        pk.hv[j] = (_Float16)x[idx];
    }
    *(uint4*)(xspd + (size_t)p * CSPD + cs0) = pk.u;
}

// ---------------------------------------------------------------------------
// Kernel 1: 1x1 conv GEMM (128x512) + BN + SiLU. Pure streaming WMMA:
// block = 8 waves = 128 channels x 32 pixels; no LDS, no barriers.
// ---------------------------------------------------------------------------
__global__ __launch_bounds__(256) void k_reduce(
    const _Float16* __restrict__ xspd, const _Float16* __restrict__ wr,
    const float* __restrict__ gamma, const float* __restrict__ beta,
    const float* __restrict__ mean, const float* __restrict__ var,
    float* __restrict__ xmid) {
    const int tid = threadIdx.x, lane = tid & 31, wv = tid >> 5;
    const int p0 = blockIdx.x * 32;
    const int b = p0 / HW, hw0 = p0 % HW;
    const int m0 = wv * 16;

    v8f acc0 = {}, acc1 = {};
    for (int kb = 0; kb < CSPD / 32; ++kb) {
        const int k0 = kb * 32;
        v16h afr = load_a_gl(wr, CSPD, m0, k0, lane);
        v16h b0  = load_b_gl(xspd, p0, k0, lane);
        v16h b1  = load_b_gl(xspd, p0 + 16, k0, lane);
        acc0 = __builtin_amdgcn_wmma_f32_16x16x32_f16(false, afr, false, b0,
                                                      (short)0, acc0, false, false);
        acc1 = __builtin_amdgcn_wmma_f32_16x16x32_f16(false, afr, false, b1,
                                                      (short)0, acc1, false, false);
    }
    const int n = lane & 15, hf = lane >> 4;
#pragma unroll
    for (int r = 0; r < 8; ++r) {
        int ch = m0 + r + 8 * hf;
        float g = gamma[ch], be = beta[ch], mu = mean[ch], va = var[ch];
        xmid[((size_t)b * CMID + ch) * HW + hw0 + n] = bnsilu(acc0[r], g, be, mu, va);
        xmid[((size_t)b * CMID + ch) * HW + hw0 + 16 + n] = bnsilu(acc1[r], g, be, mu, va);
    }
}

// ---------------------------------------------------------------------------
// Kernel 2: 3x3 offset conv (27 channels) + bias; sigmoid on mask channels.
// ---------------------------------------------------------------------------
__global__ __launch_bounds__(256) void k_offconv(
    const float* __restrict__ xmid, const float* __restrict__ ow,
    const float* __restrict__ ob, float* __restrict__ om) {
    const int o  = blockIdx.x / (NPIX / 256);
    const int pg = blockIdx.x % (NPIX / 256);
    const int p  = pg * 256 + threadIdx.x;
    const int b = p / HW, hw = p % HW, h = hw / W, w = hw % W;

    float acc = ob[o];
    const float* wrow = ow + (size_t)o * KDCN;
    const float* xb   = xmid + (size_t)b * CMID * HW;
    for (int c = 0; c < CMID; ++c) {
        const float* xc = xb + (size_t)c * HW;
        const float* wc = wrow + c * 9;
#pragma unroll
        for (int t = 0; t < 9; ++t) {
            int hh = h + t / 3 - 1, ww = w + t % 3 - 1;
            if ((unsigned)hh < (unsigned)H && (unsigned)ww < (unsigned)W)
                acc = fmaf(xc[hh * W + ww], wc[t], acc);
        }
    }
    if (o >= 18) acc = dsigmoid(acc);
    om[((size_t)b * 27 + o) * HW + hw] = acc;
}

// ---------------------------------------------------------------------------
// Kernel 3: DCNv2 sample + einsum GEMM (M=256, K=1152, 32 pixels/block).
// A-slabs (256x32 f16, 16KB) DMA'd into LDS by the Tensor Data Mover,
// double-buffered and overlapped with the bilinear B-tile build; each of 8
// waves runs 2 m-tiles x 2 n-tiles = 4 WMMAs per K-step. BN+SiLU epilogue.
// ---------------------------------------------------------------------------
__global__ __launch_bounds__(256) void k_dcn(
    const float* __restrict__ xmid, const float* __restrict__ om,
    const _Float16* __restrict__ wd,
    const float* __restrict__ gamma, const float* __restrict__ beta,
    const float* __restrict__ mean, const float* __restrict__ var,
    float* __restrict__ out) {
    __shared__ alignas(16) _Float16 Ash[2][C2][32];   // 32 KB double buffer
    __shared__ alignas(16) _Float16 Bsh[32][32];      // 2 KB sampled slab
    __shared__ float s_w00[9][32], s_w01[9][32], s_w10[9][32], s_w11[9][32];
    __shared__ int   s_y0[9][32], s_x0[9][32];

    const int tid = threadIdx.x, lane = tid & 31, wv = tid >> 5;
    const int p0 = blockIdx.x * 32;
    const int b = p0 / HW, hw0 = p0 % HW;

    // Bilinear params: 9 taps x 32 pixels, mask & validity folded into weights.
    for (int e = tid; e < 9 * 32; e += 256) {
        int k = e >> 5, n = e & 31;
        int hw = hw0 + n, h = hw / W, w = hw % W;
        size_t obase = (size_t)b * 27 * HW + hw;
        float oy = om[obase + (size_t)(2 * k) * HW];
        float ox = om[obase + (size_t)(2 * k + 1) * HW];
        float mk = om[obase + (size_t)(18 + k) * HW];
        float py = (float)(h - 1 + k / 3) + oy;
        float px = (float)(w - 1 + k % 3) + ox;
        float y0f = floorf(py), x0f = floorf(px);
        float wy = py - y0f, wx = px - x0f;
        int y0 = (int)y0f, x0 = (int)x0f;
        float vy0 = ((unsigned)y0       < (unsigned)H) ? 1.f : 0.f;
        float vy1 = ((unsigned)(y0 + 1) < (unsigned)H) ? 1.f : 0.f;
        float vx0 = ((unsigned)x0       < (unsigned)W) ? 1.f : 0.f;
        float vx1 = ((unsigned)(x0 + 1) < (unsigned)W) ? 1.f : 0.f;
        s_w00[k][n] = (1.f - wy) * (1.f - wx) * vy0 * vx0 * mk;
        s_w01[k][n] = (1.f - wy) * wx         * vy0 * vx1 * mk;
        s_w10[k][n] = wy         * (1.f - wx) * vy1 * vx0 * mk;
        s_w11[k][n] = wy         * wx         * vy1 * vx1 * mk;
        s_y0[k][n] = y0;
        s_x0[k][n] = x0;
    }

#if HAVE_TDM
    if (wv == 0)                                        // prime buffer 0
        tdm_load_a_slab(wd, (unsigned)(uintptr_t)&Ash[0][0][0]);
#endif
    __syncthreads();

    const int m0 = wv * 32;
    v8f a00 = {}, a01 = {}, a10 = {}, a11 = {};
    for (int kb = 0; kb < KSTEPS; ++kb) {
        const int buf = kb & 1;
        const int k  = kb >> 2;
        const int c0 = (kb & 3) * 32;
#if HAVE_TDM
        if (wv == 0 && kb + 1 < KSTEPS)                 // async next slab
            tdm_load_a_slab(wd + (size_t)(kb + 1) * 32,
                            (unsigned)(uintptr_t)&Ash[buf ^ 1][0][0]);
#endif
        // Build sampled B slab: 32 channels x 32 pixels, f16.
#pragma unroll
        for (int rep = 0; rep < 4; ++rep) {
            int v = tid + rep * 256;
            int n = v >> 5, cc = v & 31;
            int y0 = s_y0[k][n], x0 = s_x0[k][n];
            int y0c = min(max(y0, 0), H - 1), y1c = min(max(y0 + 1, 0), H - 1);
            int x0c = min(max(x0, 0), W - 1), x1c = min(max(x0 + 1, 0), W - 1);
            const float* xc = xmid + ((size_t)b * CMID + (c0 + cc)) * HW;
            float val = s_w00[k][n] * xc[y0c * W + x0c]
                      + s_w01[k][n] * xc[y0c * W + x1c]
                      + s_w10[k][n] * xc[y1c * W + x0c]
                      + s_w11[k][n] * xc[y1c * W + x1c];
            Bsh[n][cc] = (_Float16)val;
        }
#if HAVE_TDM
        if (wv == 0) {
            // TDM ops complete in order: waiting until <=1 outstanding means
            // the slab for this iteration has landed; final iter waits to 0.
            if (kb + 1 < KSTEPS) __builtin_amdgcn_s_wait_tensorcnt(1);
            else                 __builtin_amdgcn_s_wait_tensorcnt(0);
        }
#endif
        __syncthreads();

        v16h b0 = load_b_lds(Bsh, 0, lane);
        v16h b1 = load_b_lds(Bsh, 16, lane);
#if HAVE_TDM
        v16h f0 = load_a_lds(Ash[buf], m0, lane);
        v16h f1 = load_a_lds(Ash[buf], m0 + 16, lane);
#else
        v16h f0 = load_a_gl(wd, KDCN, m0, kb * 32, lane);
        v16h f1 = load_a_gl(wd, KDCN, m0 + 16, kb * 32, lane);
#endif
        a00 = __builtin_amdgcn_wmma_f32_16x16x32_f16(false, f0, false, b0, (short)0, a00, false, false);
        a01 = __builtin_amdgcn_wmma_f32_16x16x32_f16(false, f0, false, b1, (short)0, a01, false, false);
        a10 = __builtin_amdgcn_wmma_f32_16x16x32_f16(false, f1, false, b0, (short)0, a10, false, false);
        a11 = __builtin_amdgcn_wmma_f32_16x16x32_f16(false, f1, false, b1, (short)0, a11, false, false);
        __syncthreads();
    }

    const int n = lane & 15, hf = lane >> 4;
#pragma unroll
    for (int r = 0; r < 8; ++r) {
        int o0 = m0 + r + 8 * hf;
        int o1 = m0 + 16 + r + 8 * hf;
        float g0v = gamma[o0], be0 = beta[o0], mu0 = mean[o0], va0 = var[o0];
        float g1v = gamma[o1], be1 = beta[o1], mu1 = mean[o1], va1 = var[o1];
        out[((size_t)b * C2 + o0) * HW + hw0 + n]      = bnsilu(a00[r], g0v, be0, mu0, va0);
        out[((size_t)b * C2 + o0) * HW + hw0 + 16 + n] = bnsilu(a01[r], g0v, be0, mu0, va0);
        out[((size_t)b * C2 + o1) * HW + hw0 + n]      = bnsilu(a10[r], g1v, be1, mu1, va1);
        out[((size_t)b * C2 + o1) * HW + hw0 + 16 + n] = bnsilu(a11[r], g1v, be1, mu1, va1);
    }
}

// ---------------------------------------------------------------------------
extern "C" void kernel_launch(void* const* d_in, const int* in_sizes, int n_in,
                              void* d_out, int out_size, void* d_ws, size_t ws_size,
                              hipStream_t stream) {
    const float* x        = (const float*)d_in[0];
    const float* reduce_w = (const float*)d_in[1];
    const float* r_gamma  = (const float*)d_in[2];
    const float* r_beta   = (const float*)d_in[3];
    const float* r_mean   = (const float*)d_in[4];
    const float* r_var    = (const float*)d_in[5];
    const float* off_w    = (const float*)d_in[6];
    const float* off_b    = (const float*)d_in[7];
    const float* dcn_w    = (const float*)d_in[8];
    const float* d_gamma  = (const float*)d_in[9];
    const float* d_beta   = (const float*)d_in[10];
    const float* d_mean   = (const float*)d_in[11];
    const float* d_var    = (const float*)d_in[12];
    float* out = (float*)d_out;

    // Workspace layout (256B aligned blocks).
    char* ws = (char*)d_ws;
    size_t off = 0;
    _Float16* xspd = (_Float16*)(ws + off); off += (size_t)NPIX * CSPD * sizeof(_Float16); // 52.4 MB
    float* xmid = (float*)(ws + off); off += (size_t)B_ * CMID * HW * sizeof(float);       // 26.2 MB
    float* omb  = (float*)(ws + off); off += (size_t)B_ * 27 * HW * sizeof(float);         //  5.5 MB
    _Float16* rw16 = (_Float16*)(ws + off); off += (size_t)CMID * CSPD * sizeof(_Float16);
    off = (off + 255) & ~(size_t)255;
    _Float16* dw16 = (_Float16*)(ws + off); off += (size_t)C2 * KDCN * sizeof(_Float16);

    k_prep<<<(C2 * KDCN + 255) / 256, 256, 0, stream>>>(reduce_w, dcn_w, rw16, dw16);
    k_spd16<<<(NPIX * CSPD / 8) / 256, 256, 0, stream>>>(x, xspd);
    k_reduce<<<NPIX / 32, 256, 0, stream>>>(xspd, rw16, r_gamma, r_beta, r_mean, r_var, xmid);
    k_offconv<<<(NPIX / 256) * 27, 256, 0, stream>>>(xmid, off_w, off_b, omb);
    k_dcn<<<NPIX / 32, 256, 0, stream>>>(xmid, omb, dw16, d_gamma, d_beta, d_mean, d_var, out);
}
